// Attention_43396349559334
// MI455X (gfx1250) — matrix operand order repaired and checked by reading, not compiled, not run on hardware
//
#include <hip/hip_runtime.h>
#include <hip/hip_bf16.h>
#include <math.h>

// ---------------- problem constants (from reference) ----------------
#define BATCH  4
#define SEQ    1024
#define TOK    4096      // B*S
#define HDIM   2048      // H
#define NHq    16
#define NKVq   8
#define HD_    128
#define NQKV   4096      // (NH + 2*NKV) * HD
#define WGRP   128       // weight quant group
// --------------------------------------------------------------------

typedef _Float16 h16;
typedef __attribute__((ext_vector_type(16))) _Float16 v16h;
typedef __attribute__((ext_vector_type(8)))  _Float16 v8h;
typedef __attribute__((ext_vector_type(8)))  float    v8f;

__device__ __forceinline__ v16h make16(v8h lo, v8h hi) {
  return __builtin_shufflevector(lo, hi, 0,1,2,3,4,5,6,7,8,9,10,11,12,13,14,15);
}
// A-fragment load: per-lane row pointer p (already includes row*(ld) and k0 + hl*16),
// elements [0..7] from p, [8..15] from p+8  (contiguous k = 16*hl + s)
__device__ __forceinline__ v16h load_frag_pair(const h16* p) {
  v8h lo = *(const v8h*)p;
  v8h hi = *(const v8h*)(p + 8);
  return make16(lo, hi);
}
__device__ __forceinline__ v8f wmma_f16(v16h a, v16h b, v8f c) {
  return __builtin_amdgcn_wmma_f32_16x16x32_f16(false, a, false, b, (short)0, c, false, false);
}

// ---------------- stage 0: elementwise prep ----------------
__global__ void k_f32_to_f16(const float* __restrict__ in, h16* __restrict__ out, int n) {
  for (int i = blockIdx.x * blockDim.x + threadIdx.x; i < n; i += gridDim.x * blockDim.x)
    out[i] = (h16)in[i];
}

// wq int32 [N,2048], scale f32 [N,16]  ->  f16 [N,2048]   (K hardcoded 2048, group 128)
__global__ void k_dequant_w(const int* __restrict__ wq, const float* __restrict__ sc,
                            h16* __restrict__ out, int total) {
  for (int i = blockIdx.x * blockDim.x + threadIdx.x; i < total; i += gridDim.x * blockDim.x) {
    int n = i >> 11;
    int k = i & 2047;
    out[i] = (h16)((float)wq[i] * sc[(n << 4) + (k >> 7)]);
  }
}

// ---------------- wave-level GEMM: C[M,N] = A[M,K] * B[N,K]^T ----------------
// one wave computes a 32x64 C tile: 2 A-frags x 4 B-frags = 8 wmma per 32-deep k-step
__global__ void k_gemm_f16(const h16* __restrict__ A, const h16* __restrict__ B,
                           void* __restrict__ Cout, int M, int N, int K, int out_f32) {
  const int lane = threadIdx.x & 31;
  const int wave = threadIdx.x >> 5;
  const int tiles_n = N >> 6;                       // 64-wide N tiles
  const int tile = blockIdx.x * (blockDim.x >> 5) + wave;
  if (tile >= (M >> 5) * tiles_n) return;
  const int tm = tile / tiles_n, tn = tile % tiles_n;
  const int hl = lane >> 4, ln = lane & 15;

  const h16* Arow0 = A + (size_t)(tm * 32 + ln) * K + hl * 16;      // rows 0..15 of tile
  const h16* Arow1 = Arow0 + (size_t)16 * K;                         // rows 16..31
  const h16* Bcol = B + (size_t)(tn * 64 + ln) * K + hl * 16;       // cols (+16*K per group)

  v8f acc[2][4] = {};
  for (int k0 = 0; k0 < K; k0 += 32) {
    v16h a0 = load_frag_pair(Arow0 + k0);
    v16h a1 = load_frag_pair(Arow1 + k0);
    v16h bf[4];
    #pragma unroll
    for (int j = 0; j < 4; ++j)
      bf[j] = *(const v16h*)(Bcol + (size_t)(j * 16) * K + k0);
    #pragma unroll
    for (int j = 0; j < 4; ++j) {
      acc[0][j] = wmma_f16(a0, bf[j], acc[0][j]);
      acc[1][j] = wmma_f16(a1, bf[j], acc[1][j]);
    }
  }
  // C/D layout: lane holds col n=ln, rows r + 8*hl
  #pragma unroll
  for (int i = 0; i < 2; ++i)
    #pragma unroll
    for (int j = 0; j < 4; ++j)
      #pragma unroll
      for (int r = 0; r < 8; ++r) {
        int m = tm * 32 + i * 16 + r + 8 * hl;
        int n = tn * 64 + j * 16 + ln;
        if (out_f32) ((float*)Cout)[(size_t)m * N + n] = acc[i][j][r];
        else         ((h16*)Cout)[(size_t)m * N + n] = (h16)acc[i][j][r];
      }
}

// ---------------- RoPE + int8 round-trip quant of K/V ----------------
// xqkv f16 [TOK][32*128]; heads 0..15=Q -> rope -> qbuf [TOK][16][128]
//                         16..23=K -> rope+quantRT -> kbuf [TOK][8][128]
//                         24..31=V -> quantRT -> vt [B][8][128][SEQ] (transposed)
__global__ void k_rope_quant(const h16* __restrict__ xqkv, const int* __restrict__ start_pos,
                             h16* __restrict__ qb, h16* __restrict__ kb, h16* __restrict__ vt) {
  const int row  = blockIdx.x * 4 + (threadIdx.x >> 6);   // token*32 + head
  const int i    = threadIdx.x & 63;                      // pair index (d = 2i, 2i+1)
  const int t    = row >> 5;
  const int head = row & 31;
  const int b    = t >> 10;
  const int s    = t & 1023;

  const h16* src = xqkv + (size_t)row * 128;
  float e = (float)src[2 * i], o = (float)src[2 * i + 1];
  float r0 = e, r1 = o;
  if (head < 24) { // rope for Q and K
    float pos = (float)(start_pos[b] + s);
    float inv = __expf(-((float)(2 * i) * (1.0f / 128.0f)) * 9.210340372f); // ln(1e4)
    float sn, cs;
    __sincosf(pos * inv, &sn, &cs);
    r0 = e * cs - o * sn;
    r1 = e * sn + o * cs;
  }
  if (head < 16) {
    h16* dst = qb + ((size_t)t * NHq + head) * 128;
    dst[2 * i] = (h16)r0; dst[2 * i + 1] = (h16)r1;
  } else {
    // int8 round-trip over groups of 8 d-values = 4 consecutive threads
    float amax = fmaxf(fabsf(r0), fabsf(r1));
    amax = fmaxf(amax, __shfl_xor(amax, 1, 32));
    amax = fmaxf(amax, __shfl_xor(amax, 2, 32));
    float sc = amax * (1.0f / 127.0f) + 1e-8f;
    float q0 = fminf(fmaxf(rintf(r0 / sc), -127.f), 127.f) * sc;
    float q1 = fminf(fmaxf(rintf(r1 / sc), -127.f), 127.f) * sc;
    if (head < 24) {
      h16* dst = kb + ((size_t)t * NKVq + (head - 16)) * 128;
      dst[2 * i] = (h16)q0; dst[2 * i + 1] = (h16)q1;
    } else {
      size_t base = (((size_t)b * NKVq + (head - 24)) * 128) * SEQ;
      vt[base + (size_t)(2 * i) * SEQ + s]     = (h16)q0;
      vt[base + (size_t)(2 * i + 1) * SEQ + s] = (h16)q1;
    }
  }
}

// ---------------- flash attention ----------------
// block = 128 threads = 4 waves; wave owns 16 q rows; grid (B*NH, SEQ/64)
__global__ void k_attn(const h16* __restrict__ q, const h16* __restrict__ kk,
                       const h16* __restrict__ vt, h16* __restrict__ attn) {
  __shared__ __align__(32) h16 lds[4][16][32];
  const int lane = threadIdx.x & 31, wave = threadIdx.x >> 5;
  const int hl = lane >> 4, ln = lane & 15;
  const int b = blockIdx.x >> 4, h = blockIdx.x & 15, kvh = h >> 1;
  const int q0 = blockIdx.y * 64 + wave * 16;

  // Q fragments: 16 rows x 128 d, 4 chunks of 32
  const h16* qrow = q + (((size_t)(b * SEQ + q0 + ln)) * NHq + h) * 128;
  v16h aq[4];
  #pragma unroll
  for (int c = 0; c < 4; ++c) aq[c] = load_frag_pair(qrow + c * 32 + hl * 16);

  v8f oacc[8] = {};
  float mrow[8], lrow[8];
  #pragma unroll
  for (int j = 0; j < 8; ++j) { mrow[j] = -1e30f; lrow[j] = 0.0f; }
  const float scl = 0.08838834764831845f; // 1/sqrt(128)

  const h16* kbase = kk + ((size_t)(b * SEQ) * NKVq + kvh) * 128;
  const h16* vbase = vt + (((size_t)b * NKVq + kvh) * 128) * SEQ;

  for (int kt = 0; kt < q0 + 16; kt += 32) {
    // scores: Q(16x128) @ K^T -> 16 x 32
    v8f s0 = {}, s1 = {};
    const h16* kc0 = kbase + (size_t)(kt + ln) * (NKVq * 128) + hl * 16;
    const h16* kc1 = kbase + (size_t)(kt + 16 + ln) * (NKVq * 128) + hl * 16;
    #pragma unroll
    for (int c = 0; c < 4; ++c) {
      v16h b0 = *(const v16h*)(kc0 + c * 32);
      v16h b1 = *(const v16h*)(kc1 + c * 32);
      s0 = wmma_f16(aq[c], b0, s0);
      s1 = wmma_f16(aq[c], b1, s1);
    }
    // online softmax; lane holds cols kt+ln / kt+16+ln for rows j+8*hl
    #pragma unroll
    for (int j = 0; j < 8; ++j) {
      int rowg = q0 + j + 8 * hl;
      float v0 = s0[j] * scl; if (kt + ln > rowg)      v0 = -1e30f;
      float v1 = s1[j] * scl; if (kt + 16 + ln > rowg) v1 = -1e30f;
      float tmax = fmaxf(v0, v1);
      tmax = fmaxf(tmax, __shfl_xor(tmax, 1, 32));
      tmax = fmaxf(tmax, __shfl_xor(tmax, 2, 32));
      tmax = fmaxf(tmax, __shfl_xor(tmax, 4, 32));
      tmax = fmaxf(tmax, __shfl_xor(tmax, 8, 32));
      float mnew = fmaxf(mrow[j], tmax);
      float corr = __expf(mrow[j] - mnew);
      float p0 = __expf(v0 - mnew), p1 = __expf(v1 - mnew);
      float rs = p0 + p1;
      rs += __shfl_xor(rs, 1, 32);
      rs += __shfl_xor(rs, 2, 32);
      rs += __shfl_xor(rs, 4, 32);
      rs += __shfl_xor(rs, 8, 32);
      lrow[j] = lrow[j] * corr + rs;
      mrow[j] = mnew;
      #pragma unroll
      for (int c = 0; c < 8; ++c) oacc[c][j] *= corr;
      // P in C-layout -> LDS (row j+8*hl, cols ln / 16+ln)
      lds[wave][j + 8 * hl][ln]      = (h16)p0;
      lds[wave][j + 8 * hl][16 + ln] = (h16)p1;
    }
    __builtin_amdgcn_wave_barrier();
    // reload P as A-fragment (16x32): lane row=ln, cols hl*16..hl*16+15
    v16h ap = load_frag_pair(&lds[wave][ln][hl * 16]);
    __builtin_amdgcn_wave_barrier();
    // PV: P(16x32) @ V(32x128); V transposed so B frags are contiguous
    #pragma unroll
    for (int c = 0; c < 8; ++c) {
      v16h bv = *(const v16h*)(vbase + (size_t)(c * 16 + ln) * SEQ + kt + hl * 16);
      oacc[c] = wmma_f16(ap, bv, oacc[c]);
    }
    __builtin_amdgcn_wave_barrier();
  }
  // epilogue: divide by l, write attn [TOK][NH*HD] f16
  #pragma unroll
  for (int c = 0; c < 8; ++c)
    #pragma unroll
    for (int j = 0; j < 8; ++j) {
      float ov = oacc[c][j] / lrow[j];
      int rowg = q0 + j + 8 * hl;
      attn[((size_t)(b * SEQ + rowg)) * (NHq * 128) + h * 128 + c * 16 + ln] = (h16)ov;
    }
}

// ---------------- launcher ----------------
extern "C" void kernel_launch(void* const* d_in, const int* in_sizes, int n_in,
                              void* d_out, int out_size, void* d_ws, size_t ws_size,
                              hipStream_t stream) {
  const float* x      = (const float*)d_in[0];
  const int*   wqkv_q = (const int*)  d_in[1];
  const float* wqkv_s = (const float*)d_in[2];
  const int*   wo_q   = (const int*)  d_in[3];
  const float* wo_s   = (const float*)d_in[4];
  const int*   start_pos = (const int*)d_in[8];

  char* ws = (char*)d_ws;
  h16* x_h    = (h16*)(ws + 0);           // 16 MB  [TOK][HDIM]
  h16* wqkv_h = (h16*)(ws + 16777216);    // 16 MB  [NQKV][HDIM]
  h16* wo_h   = (h16*)(ws + 33554432);    //  8 MB  [HDIM][HDIM]
  h16* xqkv   = (h16*)(ws + 41943040);    // 32 MB  [TOK][NQKV]
  h16* qbuf   = (h16*)(ws + 75497472);    // 16 MB  [TOK][16][128]
  h16* kbuf   = (h16*)(ws + 92274688);    //  8 MB  [TOK][8][128]
  h16* vtbuf  = (h16*)(ws + 100663296);   //  8 MB  [B][8][128][SEQ]
  h16* attnb  = (h16*)(ws + 109051904);   // 16 MB  [TOK][2048]

  k_f32_to_f16<<<2048, 256, 0, stream>>>(x, x_h, TOK * HDIM);
  k_dequant_w <<<2048, 256, 0, stream>>>(wqkv_q, wqkv_s, wqkv_h, NQKV * HDIM);
  k_dequant_w <<<2048, 256, 0, stream>>>(wo_q, wo_s, wo_h, HDIM * HDIM);

  // QKV GEMM: 4096x4096x2048, (128 m-tiles x 64 n-tiles)=8192 tiles, 8 waves/block
  k_gemm_f16<<<1024, 256, 0, stream>>>(x_h, wqkv_h, xqkv, TOK, NQKV, HDIM, 0);

  // RoPE + KV quant round-trip: TOK*32 rows, 4 rows per 256-thread block
  k_rope_quant<<<TOK * 32 / 4, 256, 0, stream>>>(xqkv, start_pos, qbuf, kbuf, vtbuf);

  // flash attention: (B*NH) x (SEQ/64) blocks of 4 waves
  k_attn<<<dim3(BATCH * NHq, SEQ / 64), 128, 0, stream>>>(qbuf, kbuf, vtbuf, attnb);

  // output GEMM: 4096x2048x2048 -> f32 d_out, (128 x 32)=4096 tiles, 8 waves/block
  k_gemm_f16<<<512, 256, 0, stream>>>(attnb, wo_h, d_out, TOK, HDIM, HDIM, 1);
}
